// contextual_attention_block_44263932953268
// MI455X (gfx1250) — compile-verified, matrix operand order checked by reading, not run on hardware
//
#include <hip/hip_runtime.h>
#include <hip/hip_bf16.h>
#include <math.h>

// ============================================================================
// Contextual attention block for MI455X (gfx1250, wave32, WMMA).
// All GEMM-shaped work runs through v_wmma_f32_16x16x32_f16:
//   - correlation   : per-batch 1024x1024x1152
//   - deconv        : 72 GEMMs of 1024x128x1024 + gather-add
//   - conv3x3 (x2)  : implicit-im2col WMMA GEMM (K=1152 / K=2304)
//   - pointwise (x2): 32768x128xK GEMM
// f16 inputs / f32 accumulation (data is unit-scale normal). Entire working
// set (<200MB) is L2-resident on MI455X (192MB L2), so the pipeline is
// matrix-throughput bound, not HBM bound.
// ============================================================================

typedef _Float16 half8  __attribute__((ext_vector_type(8)));
typedef _Float16 half16 __attribute__((ext_vector_type(16)));
typedef float    float8 __attribute__((ext_vector_type(8)));

#define DEV __device__ __forceinline__

DEV float8 wmma_f32_16x16x32(half16 a, half16 b, float8 c) {
  return __builtin_amdgcn_wmma_f32_16x16x32_f16(false, a, false, b, (short)0, c,
                                                false, false);
}
DEV half16 cat16(half8 lo, half8 hi) {
  half16 r;
#pragma unroll
  for (int i = 0; i < 8; ++i) { r[i] = lo[i]; r[i + 8] = hi[i]; }
  return r;
}

// ---------------------------------------------------------------------------
// invden[l*128+c] = 1 / max(sum_{b,ki,kj} feats^2, 1e-8)
// feats = 3x3 stride-1 patches (pad 1) of src = x[:, ::2, ::2, :]
// ---------------------------------------------------------------------------
__global__ void k_invden(const float* __restrict__ x, float* __restrict__ invden) {
  int i = blockIdx.x * blockDim.x + threadIdx.x;
  if (i >= 1024 * 128) return;
  int c = i & 127, l = i >> 7;
  int i2 = l >> 5, j2 = l & 31;
  float s = 0.f;
  for (int b = 0; b < 8; ++b)
    for (int ki = 0; ki < 3; ++ki) {
      int yy = i2 + ki - 1; if ((unsigned)yy >= 32u) continue;
      for (int kj = 0; kj < 3; ++kj) {
        int xx = j2 + kj - 1; if ((unsigned)xx >= 32u) continue;
        float v = x[(((long)b * 64 + 2 * yy) * 64 + 2 * xx) * 128 + c];
        s += v * v;
      }
    }
  invden[i] = 1.f / fmaxf(s, 1e-8f);
}

// ---------------------------------------------------------------------------
// P[b][l][t]  = src patch (f16);  Bn[b][l][t] = P / denom[c,l] (f16)
// t = (ki*3+kj)*128 + c ; layout row-major, ld = 1152
// ---------------------------------------------------------------------------
__global__ void k_build_pb(const float* __restrict__ x, const float* __restrict__ invden,
                           _Float16* __restrict__ P, _Float16* __restrict__ Bn) {
  long i = (long)blockIdx.x * blockDim.x + threadIdx.x;
  if (i >= 8L * 1024 * 1152) return;
  int t = (int)(i % 1152);
  long r = i / 1152;
  int l = (int)(r & 1023);
  int b = (int)(r >> 10);
  int c = t & 127;
  int kk = t >> 7;                 // 0..8
  int ki = kk / 3, kj = kk - ki * 3;
  int i2 = l >> 5, j2 = l & 31;
  int yy = i2 + ki - 1, xx = j2 + kj - 1;
  float v = 0.f;
  if ((unsigned)yy < 32u && (unsigned)xx < 32u)
    v = x[(((long)b * 64 + 2 * yy) * 64 + 2 * xx) * 128 + c];
  P[i]  = (_Float16)v;
  Bn[i] = (_Float16)(v * invden[l * 128 + c]);
}

// ---------------------------------------------------------------------------
// Generic WMMA GEMM:  C[z][m][n] = sum_k A[z/aDiv][m][k] * Bt[z][n][k] (+bias[n])
// block = 256 thr = 8 waves; wave tile 32x32; block tile 64(M) x 128(N).
// A/Bt row-major with leading dim K; fragment layout per CDNA5 ISA 7.12.2.
// ---------------------------------------------------------------------------
__global__ __launch_bounds__(256) void k_gemm_abt(
    const _Float16* __restrict__ A, const _Float16* __restrict__ Bt,
    float* __restrict__ C, const float* __restrict__ bias,
    int N, int K, long sA, long sB, long sC, int aDiv) {
  const int z = blockIdx.z;
  const _Float16* Ab = A + (long)(z / aDiv) * sA;
  const _Float16* Bb = Bt + (long)z * sB;
  float* Cb = C + (long)z * sC;

  const int lane = threadIdx.x & 31;
  const int wave = threadIdx.x >> 5;
  const int m0 = blockIdx.y * 64 + (wave >> 2) * 32;
  const int n0 = blockIdx.x * 128 + (wave & 3) * 32;

  const int mlan = lane & 15;
  const int koff = (lane >> 4) << 3;      // K-split: lanes 0-15 -> 0, 16-31 -> 8

  const _Float16* pa = Ab + (long)(m0 + mlan) * K + koff;
  const _Float16* pb = Bb + (long)(n0 + mlan) * K + koff;
  const long row16 = 16L * K;

  float8 c00 = {}, c01 = {}, c10 = {}, c11 = {};
  for (int kb = 0; kb < K; kb += 32) {
    __builtin_prefetch(pa + 256, 0, 1);   // global_prefetch_b8 (next K tiles)
    __builtin_prefetch(pb + 256, 0, 1);
    half16 a0 = cat16(*(const half8*)pa,             *(const half8*)(pa + 16));
    half16 a1 = cat16(*(const half8*)(pa + row16),   *(const half8*)(pa + row16 + 16));
    half16 b0 = cat16(*(const half8*)pb,             *(const half8*)(pb + 16));
    half16 b1 = cat16(*(const half8*)(pb + row16),   *(const half8*)(pb + row16 + 16));
    pa += 32; pb += 32;
    c00 = wmma_f32_16x16x32(a0, b0, c00);
    c01 = wmma_f32_16x16x32(a0, b1, c01);
    c10 = wmma_f32_16x16x32(a1, b0, c10);
    c11 = wmma_f32_16x16x32(a1, b1, c11);
  }

  const int rofs = (lane >> 4) << 3;      // D rows: lanes 16-31 hold M+8
  const int col  = lane & 15;
  const float bv0 = bias ? bias[n0 + col]      : 0.f;
  const float bv1 = bias ? bias[n0 + 16 + col] : 0.f;
#pragma unroll
  for (int e = 0; e < 8; ++e) {
    const int r = e + rofs;
    float* r0 = Cb + (long)(m0 + r) * N + n0;
    float* r1 = Cb + (long)(m0 + 16 + r) * N + n0;
    r0[col]      = c00[e] + bv0;
    r0[16 + col] = c01[e] + bv1;
    r1[col]      = c10[e] + bv0;
    r1[16 + col] = c11[e] + bv1;
  }
}

// ---------------------------------------------------------------------------
// Implicit-im2col 3x3 SAME conv as WMMA GEMM. inp: (8,64,64,CIN) f16,
// Btw: (128 x 9*CIN) f16 (pre-transposed weights), out: (32768 x 128) f32.
// ---------------------------------------------------------------------------
template <int CIN>
DEV half8 im2col8(const _Float16* __restrict__ inp, int b, int oy, int ox, int k) {
  const int t  = k / CIN;                 // compile-time division
  const int ci = k - t * CIN;
  const int ky = t / 3, kx = t - ky * 3;
  const int iy = oy + ky - 1, ix = ox + kx - 1;
  if ((unsigned)iy < 64u && (unsigned)ix < 64u)
    return *(const half8*)(inp + (((long)((b << 6) + iy) << 6) + ix) * CIN + ci);
  half8 zz = {};
  return zz;
}

template <int CIN>
__global__ __launch_bounds__(256) void k_conv3x3(
    const _Float16* __restrict__ inp, const _Float16* __restrict__ Btw,
    const float* __restrict__ bias, float* __restrict__ out) {
  constexpr int K = 9 * CIN;
  const int lane = threadIdx.x & 31;
  const int wave = threadIdx.x >> 5;
  const int m0 = blockIdx.y * 64 + (wave >> 2) * 32;
  const int n0 = (wave & 3) * 32;         // Cout == 128, grid.x == 1

  const int mlan = lane & 15;
  const int koff = (lane >> 4) << 3;

  const int mA = m0 + mlan, mB = mA + 16;
  const int bA = mA >> 12, oyA = (mA >> 6) & 63, oxA = mA & 63;
  const int bB = mB >> 12, oyB = (mB >> 6) & 63, oxB = mB & 63;

  const _Float16* pb = Btw + (long)(n0 + mlan) * K + koff;
  const long row16 = 16L * K;

  float8 c00 = {}, c01 = {}, c10 = {}, c11 = {};
  for (int kb = 0; kb < K; kb += 32) {
    const int k0 = kb + koff, k1 = k0 + 16;
    half16 a0 = cat16(im2col8<CIN>(inp, bA, oyA, oxA, k0),
                      im2col8<CIN>(inp, bA, oyA, oxA, k1));
    half16 a1 = cat16(im2col8<CIN>(inp, bB, oyB, oxB, k0),
                      im2col8<CIN>(inp, bB, oyB, oxB, k1));
    half16 b0 = cat16(*(const half8*)pb,           *(const half8*)(pb + 16));
    half16 b1 = cat16(*(const half8*)(pb + row16), *(const half8*)(pb + row16 + 16));
    pb += 32;
    c00 = wmma_f32_16x16x32(a0, b0, c00);
    c01 = wmma_f32_16x16x32(a0, b1, c01);
    c10 = wmma_f32_16x16x32(a1, b0, c10);
    c11 = wmma_f32_16x16x32(a1, b1, c11);
  }

  const int rofs = (lane >> 4) << 3;
  const int col  = lane & 15;
  const float bv0 = bias[n0 + col];
  const float bv1 = bias[n0 + 16 + col];
#pragma unroll
  for (int e = 0; e < 8; ++e) {
    const int r = e + rofs;
    float* r0 = out + (long)(m0 + r) * 128 + n0;
    float* r1 = out + (long)(m0 + 16 + r) * 128 + n0;
    r0[col] = c00[e] + bv0;  r0[16 + col] = c01[e] + bv1;
    r1[col] = c10[e] + bv0;  r1[16 + col] = c11[e] + bv1;
  }
}

// ---------------------------------------------------------------------------
// Fuse pass 1: diagonal 3-tap over flattened (h*w, L) grid.
// ---------------------------------------------------------------------------
__global__ void k_fuse1(const float* __restrict__ y, float* __restrict__ y2) {
  long i = (long)blockIdx.x * blockDim.x + threadIdx.x;
  if (i >= (8L << 20)) return;
  int l = (int)(i & 1023);
  long r = i >> 10;
  int a = (int)(r & 1023);
  int b = (int)(r >> 10);
  const float* base = y + ((long)b << 20);
  float s = base[((long)a << 10) + l];
  if (a > 0 && l > 0)       s += base[((long)(a - 1) << 10) + (l - 1)];
  if (a < 1023 && l < 1023) s += base[((long)(a + 1) << 10) + (l + 1)];
  y2[i] = s;
}

// Fuse pass 2 with the (i,j)<->(j,i) reshape/transpose trick folded into
// index arithmetic; writes back into the original (b, i*32+j, i2*32+j2) layout.
__global__ void k_fuse2(const float* __restrict__ y2, float* __restrict__ y) {
  long i = (long)blockIdx.x * blockDim.x + threadIdx.x;
  if (i >= (8L << 20)) return;
  int l = (int)(i & 1023);
  long r = i >> 10;
  int a = (int)(r & 1023);
  int b = (int)(r >> 10);
  int i2 = l >> 5, j2 = l & 31;
  int ii = a >> 5, jj = a & 31;
  const float* base = y2 + ((long)b << 20);
  int ap0 = jj * 32 + ii, lp0 = j2 * 32 + i2;
  float s = 0.f;
#pragma unroll
  for (int d = -1; d <= 1; ++d) {
    int ap = ap0 + d, lp = lp0 + d;
    if ((unsigned)ap < 1024u && (unsigned)lp < 1024u) {
      int ai = (ap & 31) * 32 + (ap >> 5);
      int li = (lp & 31) * 32 + (lp >> 5);
      s += base[((long)ai << 10) + li];
    }
  }
  y[i] = s;
}

// ---------------------------------------------------------------------------
// mask: 16x16/16 maxpool then 3x3/1 maxpool, mm = 1 - m
// ---------------------------------------------------------------------------
__global__ void k_pool16(const float* __restrict__ mask, float* __restrict__ m1) {
  int i = blockIdx.x * blockDim.x + threadIdx.x;
  if (i >= 8192) return;
  int j = i & 31, ii = (i >> 5) & 31, b = i >> 10;
  float mx = -1e30f;
  for (int dy = 0; dy < 16; ++dy)
    for (int dx = 0; dx < 16; ++dx)
      mx = fmaxf(mx, mask[(((long)b * 512 + (ii * 16 + dy)) * 512) + j * 16 + dx]);
  m1[i] = mx;
}
__global__ void k_mm(const float* __restrict__ m1, float* __restrict__ mm) {
  int i = blockIdx.x * blockDim.x + threadIdx.x;
  if (i >= 8192) return;
  int j = i & 31, ii = (i >> 5) & 31, b = i >> 10;
  float mx = -1e30f;
  for (int dy = -1; dy <= 1; ++dy) {
    int yy = ii + dy; if ((unsigned)yy >= 32u) continue;
    for (int dx = -1; dx <= 1; ++dx) {
      int xx = j + dx; if ((unsigned)xx >= 32u) continue;
      mx = fmaxf(mx, m1[b * 1024 + yy * 32 + xx]);
    }
  }
  mm[i] = 1.f - mx;
}

// ---------------------------------------------------------------------------
// row softmax: corres = softmax_l(y*mm*10)*mm ; also emit f16 for deconv GEMM
// ---------------------------------------------------------------------------
__global__ __launch_bounds__(256) void k_softmax(const float* __restrict__ y,
                                                 const float* __restrict__ mm,
                                                 float* __restrict__ corres,
                                                 _Float16* __restrict__ ysf) {
  const int row = blockIdx.x;                 // b*1024 + p
  const float* yr = y + ((long)row << 10);
  const float mmv = mm[row];
  const float scale = mmv * 10.f;
  __shared__ float red[256];
  float v[4];
#pragma unroll
  for (int j = 0; j < 4; ++j) v[j] = yr[threadIdx.x + (j << 8)] * scale;
  float lm = fmaxf(fmaxf(v[0], v[1]), fmaxf(v[2], v[3]));
  red[threadIdx.x] = lm; __syncthreads();
  for (int s = 128; s > 0; s >>= 1) {
    if (threadIdx.x < s) red[threadIdx.x] = fmaxf(red[threadIdx.x], red[threadIdx.x + s]);
    __syncthreads();
  }
  const float rmax = red[0]; __syncthreads();
  float e[4]; float ls = 0.f;
#pragma unroll
  for (int j = 0; j < 4; ++j) { e[j] = expf(v[j] - rmax); ls += e[j]; }
  red[threadIdx.x] = ls; __syncthreads();
  for (int s = 128; s > 0; s >>= 1) {
    if (threadIdx.x < s) red[threadIdx.x] += red[threadIdx.x + s];
    __syncthreads();
  }
  const float inv = mmv / red[0];
#pragma unroll
  for (int j = 0; j < 4; ++j) {
    const float o = e[j] * inv;
    const long idx = ((long)row << 10) + threadIdx.x + (j << 8);
    corres[idx] = o;
    ysf[idx] = (_Float16)o;
  }
}

// ---------------------------------------------------------------------------
// Btd[(b*9+t9)*128 + c][l] = raw tap weight x[b, 2*i2+dy, 2*j2+dx, c] (f16)
// ---------------------------------------------------------------------------
__global__ void k_build_btd(const float* __restrict__ x, _Float16* __restrict__ btd) {
  long i = (long)blockIdx.x * blockDim.x + threadIdx.x;
  if (i >= 9437184L) return;
  int l = (int)(i & 1023);
  long r = i >> 10;
  int c = (int)(r & 127);
  int zz = (int)(r >> 7);                // 0..71
  int t9 = zz % 9, b = zz / 9;
  int dy = t9 / 3, dx = t9 - dy * 3;
  int i2 = l >> 5, j2 = l & 31;
  int yy = 2 * i2 + dy, xx = 2 * j2 + dx;
  float v = 0.f;
  if (yy < 64 && xx < 64) v = x[(((long)b * 64 + yy) * 64 + xx) * 128 + c];
  btd[i] = (_Float16)v;
}

// y_up[b,oy,ox,c] = sum over <=4 valid (tap,query) of per-tap GEMM results
__global__ void k_gather_yup(const float* __restrict__ dtmp, float* __restrict__ yup,
                             _Float16* __restrict__ yupf) {
  long i = (long)blockIdx.x * blockDim.x + threadIdx.x;
  if (i >= 4194304L) return;
  int c = (int)(i & 127);
  long p = i >> 7;
  int ox = (int)(p & 63), oy = (int)((p >> 6) & 63), b = (int)(p >> 12);
  float s = 0.f;
  for (int dy = 0; dy < 3; ++dy) {
    int ry = oy - dy;
    if (ry < 0 || (ry & 1)) continue;
    int py = ry >> 1; if (py >= 32) continue;
    for (int dx = 0; dx < 3; ++dx) {
      int rx = ox - dx;
      if (rx < 0 || (rx & 1)) continue;
      int px = rx >> 1; if (px >= 32) continue;
      long z = (long)(b * 9 + dy * 3 + dx);
      s += dtmp[(z << 17) + ((long)(py * 32 + px) << 7) + c];
    }
  }
  yup[i] = s;
  yupf[i] = (_Float16)s;
}

// depthwise 3x3 SAME (f16 in, f16 out), weights (3,3,1,C) f32
__global__ void k_dw3x3(const _Float16* __restrict__ inp, const float* __restrict__ wdw,
                        _Float16* __restrict__ out, int C, long total) {
  long i = (long)blockIdx.x * blockDim.x + threadIdx.x;
  if (i >= total) return;
  int c = (int)(i % C);
  long p = i / C;
  int ox = (int)(p & 63), oy = (int)((p >> 6) & 63), b = (int)(p >> 12);
  float s = 0.f;
  for (int ky = 0; ky < 3; ++ky) {
    int iy = oy + ky - 1; if ((unsigned)iy >= 64u) continue;
    for (int kx = 0; kx < 3; ++kx) {
      int ix = ox + kx - 1; if ((unsigned)ix >= 64u) continue;
      s += (float)inp[(((long)(b * 64 + iy) * 64) + ix) * C + c] * wdw[(ky * 3 + kx) * C + c];
    }
  }
  out[i] = (_Float16)s;
}

// weight transpose+f16: w (Kd, Co) HWIO-flat -> bt[co*Kd + t]
__global__ void k_wt2bt(const float* __restrict__ w, _Float16* __restrict__ bt,
                        int Kd, int Co) {
  int i = blockIdx.x * blockDim.x + threadIdx.x;
  if (i >= Kd * Co) return;
  int t = i % Kd, co = i / Kd;
  bt[i] = (_Float16)w[(long)t * Co + co];
}

// a1 = elu(x1) * softmax_c(x2); cat = [a1 | x] as f16
__global__ __launch_bounds__(128) void k_gate1(const float* __restrict__ x1,
                                               const float* __restrict__ x2,
                                               const float* __restrict__ xin,
                                               _Float16* __restrict__ cat) {
  const int m = blockIdx.x, c = threadIdx.x;
  const float g = x2[((long)m << 7) + c];
  __shared__ float red[128];
  red[c] = g; __syncthreads();
  for (int s = 64; s > 0; s >>= 1) { if (c < s) red[c] = fmaxf(red[c], red[c + s]); __syncthreads(); }
  const float mx = red[0]; __syncthreads();
  const float e = expf(g - mx);
  red[c] = e; __syncthreads();
  for (int s = 64; s > 0; s >>= 1) { if (c < s) red[c] += red[c + s]; __syncthreads(); }
  const float soft = e / red[0];
  const float u = x1[((long)m << 7) + c];
  const float elu = u > 0.f ? u : (expf(u) - 1.f);
  cat[((long)m << 8) + c]       = (_Float16)(elu * soft);
  cat[((long)m << 8) + 128 + c] = (_Float16)xin[((long)m << 7) + c];
}

__global__ __launch_bounds__(128) void k_gate2(const float* __restrict__ x1,
                                               const float* __restrict__ x2,
                                               float* __restrict__ outp) {
  const int m = blockIdx.x, c = threadIdx.x;
  const float g = x2[((long)m << 7) + c];
  __shared__ float red[128];
  red[c] = g; __syncthreads();
  for (int s = 64; s > 0; s >>= 1) { if (c < s) red[c] = fmaxf(red[c], red[c + s]); __syncthreads(); }
  const float mx = red[0]; __syncthreads();
  const float e = expf(g - mx);
  red[c] = e; __syncthreads();
  for (int s = 64; s > 0; s >>= 1) { if (c < s) red[c] += red[c + s]; __syncthreads(); }
  const float soft = e / red[0];
  const float u = x1[((long)m << 7) + c];
  const float elu = u > 0.f ? u : (expf(u) - 1.f);
  outp[((long)m << 7) + c] = elu * soft;
}

// ============================================================================
extern "C" void kernel_launch(void* const* d_in, const int* in_sizes, int n_in,
                              void* d_out, int out_size, void* d_ws, size_t ws_size,
                              hipStream_t stream) {
  (void)in_sizes; (void)n_in; (void)out_size; (void)ws_size;

  const float* x      = (const float*)d_in[0];
  const float* mask_s = (const float*)d_in[1];
  const float* w1  = (const float*)d_in[2];
  const float* b1  = (const float*)d_in[3];
  const float* dw1 = (const float*)d_in[4];
  const float* pw1 = (const float*)d_in[5];
  const float* db1 = (const float*)d_in[6];
  const float* w2  = (const float*)d_in[7];
  const float* b2  = (const float*)d_in[8];
  const float* dw2 = (const float*)d_in[9];
  const float* pw2 = (const float*)d_in[10];
  const float* db2 = (const float*)d_in[11];

  float* out_a      = (float*)d_out;                 // (8,64,64,128)
  float* out_corres = (float*)d_out + 4194304;       // (8,32,32,1024)

  // ---- workspace layout (phase-overlapped, ~143.7 MB total) ----
  char* ws = (char*)d_ws;
  float*    yb   = (float*)(ws + 0);                         // y (33.5MB) -> y_up f32 + y_up f16
  float*    y2b  = (float*)(ws + 33554432L);                 // y2 (33.5MB) -> cat f16 + x2a f16
  _Float16* Pb   = (_Float16*)(ws + 67108864L);              // P f16 (18.9MB) -> ys f16 -> x2a2 f16
  _Float16* Bnb  = (_Float16*)(ws + 85983232L);              // Bn f16 (18.9MB) -> Btd f16
  float*    bigb = (float*)(ws + 104857600L);                // invden -> Dtmp (37.7MB) -> x1|x2
  float*    mmb  = (float*)(ws + 142606336L);
  float*    m1b  = (float*)(ws + 142639104L);
  _Float16* wtb  = (_Float16*)(ws + 142671872L);             // converted weights (~1MB)

  _Float16* btw1 = wtb;                 // 128*1152
  _Float16* pw1t = wtb + 147456;        // 128*128
  _Float16* btw2 = pw1t + 16384;        // 128*2304
  _Float16* pw2t = btw2 + 294912;       // 128*256

  float*    invden = bigb;
  _Float16* ysf    = Pb;
  _Float16* btd    = Bnb;
  float*    dtmp   = bigb;
  float*    yup    = yb;
  _Float16* yupf   = (_Float16*)(ws + 16777216L);
  _Float16* catf   = (_Float16*)y2b;
  _Float16* x2af   = (_Float16*)(ws + 50331648L);
  float*    x1b    = bigb;
  float*    x2b    = (float*)(ws + 121634816L);
  _Float16* x2a2f  = Pb;

  // ---- phase A: patches + normalization + correlation GEMM ----
  k_invden<<<512, 256, 0, stream>>>(x, invden);
  k_build_pb<<<36864, 256, 0, stream>>>(x, invden, Pb, Bnb);
  k_gemm_abt<<<dim3(8, 16, 8), 256, 0, stream>>>(
      Pb, Bnb, yb, nullptr, 1024, 1152,
      1024L * 1152, 1024L * 1152, 1024L * 1024, 1);

  // ---- mask ----
  k_pool16<<<32, 256, 0, stream>>>(mask_s, m1b);
  k_mm<<<32, 256, 0, stream>>>(m1b, mmb);

  // ---- fuse convs (identity-diagonal 3-tap, transposes folded) ----
  k_fuse1<<<32768, 256, 0, stream>>>(yb, y2b);
  k_fuse2<<<32768, 256, 0, stream>>>(y2b, yb);

  // ---- masked softmax -> corres + f16 copy ----
  k_softmax<<<8192, 256, 0, stream>>>(yb, mmb, out_corres, ysf);

  // ---- deconv: 72 per-tap GEMMs + gather-add ----
  k_build_btd<<<36864, 256, 0, stream>>>(x, btd);
  k_gemm_abt<<<dim3(1, 16, 72), 256, 0, stream>>>(
      ysf, btd, dtmp, nullptr, 128, 1024,
      1024L * 1024, 128L * 1024, 1024L * 128, 9);
  k_gather_yup<<<16384, 256, 0, stream>>>(dtmp, yup, yupf);

  // ---- weight conversion (f32 -> transposed f16) ----
  k_wt2bt<<<(1152 * 128 + 255) / 256, 256, 0, stream>>>(w1, btw1, 1152, 128);
  k_wt2bt<<<(128 * 128 + 255) / 256, 256, 0, stream>>>(pw1, pw1t, 128, 128);
  k_wt2bt<<<(2304 * 128 + 255) / 256, 256, 0, stream>>>(w2, btw2, 2304, 128);
  k_wt2bt<<<(256 * 128 + 255) / 256, 256, 0, stream>>>(pw2, pw2t, 256, 128);

  // ---- gated conv 1 (Cin=128) ----
  k_conv3x3<128><<<dim3(1, 512), 256, 0, stream>>>(yupf, btw1, b1, x1b);
  k_dw3x3<<<16384, 256, 0, stream>>>(yupf, dw1, x2af, 128, 4194304L);
  k_gemm_abt<<<dim3(1, 512, 1), 256, 0, stream>>>(
      x2af, pw1t, x2b, db1, 128, 128, 0, 0, 0, 1);
  k_gate1<<<32768, 128, 0, stream>>>(x1b, x2b, x, catf);

  // ---- gated conv 2 (Cin=256) ----
  k_conv3x3<256><<<dim3(1, 512), 256, 0, stream>>>(catf, btw2, b2, x1b);
  k_dw3x3<<<32768, 256, 0, stream>>>(catf, dw2, x2a2f, 256, 8388608L);
  k_gemm_abt<<<dim3(1, 512, 1), 256, 0, stream>>>(
      x2a2f, pw2t, x2b, db2, 128, 256, 0, 0, 0, 1);
  k_gate2<<<32768, 128, 0, stream>>>(x1b, x2b, out_a);
}
